// MultiHeadAttention_34419867910412
// MI455X (gfx1250) — matrix operand order repaired and checked by reading, not compiled, not run on hardware
//
#include <hip/hip_runtime.h>
#include <hip/hip_bf16.h>
#include <stdint.h>

// Problem constants (reference: B=2, S=2048, D=1024, H=16, E=64)
#define B_ 2
#define S_ 2048
#define D_ 1024
#define H_ 16
#define E_ 64

typedef __attribute__((ext_vector_type(16))) __bf16 bf16x16;
typedef __attribute__((ext_vector_type(8)))  __bf16 bf16x8;
typedef __attribute__((ext_vector_type(8)))  float  f32x8;

#define WMMA_BF16(A, Bm, C) \
  __builtin_amdgcn_wmma_f32_16x16x32_bf16(false, (A), false, (Bm), (short)0, (C), false, false)

// Native f32 -> bf16 conversion (LLVM lowers with RNE; enables v_cvt_pk_bf16_f32)
static __device__ __forceinline__ __bf16 tobf(float f) { return (__bf16)f; }

static __device__ __forceinline__ f32x8 zero8() {
  f32x8 z;
#pragma unroll
  for (int i = 0; i < 8; ++i) z[i] = 0.0f;
  return z;
}

// A-matrix fragment (16x32 bf16, MxK): lane holds row m=lane%16, g=lane/16.
// halves 0..7  -> k = kb + 16g + i
// halves 8..15 -> k = kb + 16g + 8 + (i-8)
static __device__ __forceinline__ bf16x16 a_frag(const __bf16* __restrict__ rowp, int kb, int g) {
  bf16x8 lo = *(const bf16x8*)(rowp + kb + 16 * g);
  bf16x8 hi = *(const bf16x8*)(rowp + kb + 16 * g + 8);
  bf16x16 r;
#pragma unroll
  for (int i = 0; i < 8; ++i) { r[i] = lo[i]; r[i + 8] = hi[i]; }
  return r;
}

// ---------------------------------------------------------------------------
// Conversions
// ---------------------------------------------------------------------------
__global__ void cvt_x_kernel(const float* __restrict__ in, __bf16* __restrict__ out, int n) {
  int i = blockIdx.x * blockDim.x + threadIdx.x;
  if (i < n) out[i] = tobf(in[i]);
}

// Wq/Wk/Wv: [H,D,E] f32 -> transposed [H,E,D] bf16 (Bᵀ rows contiguous in D)
__global__ void cvt_w_kernel(const float* __restrict__ wq, const float* __restrict__ wk,
                             const float* __restrict__ wv, __bf16* __restrict__ wqT,
                             __bf16* __restrict__ wkT, __bf16* __restrict__ wvT) {
  int i = blockIdx.x * blockDim.x + threadIdx.x;
  const int n = H_ * D_ * E_;
  if (i >= n) return;
  int e = i % E_;
  int d = (i / E_) % D_;
  int h = i / (D_ * E_);
  size_t o = (size_t)h * E_ * D_ + (size_t)e * D_ + d;
  wqT[o] = tobf(wq[i]);
  wkT[o] = tobf(wk[i]);
  wvT[o] = tobf(wv[i]);
}

__global__ void cvt_wp_kernel(const float* __restrict__ in, __bf16* __restrict__ out, int n) {
  int i = blockIdx.x * blockDim.x + threadIdx.x;
  if (i < n) out[i] = tobf(in[i]);  // Wproj [D,D] row-major == Bᵀ layout for y = a Wᵀ
}

// ---------------------------------------------------------------------------
// Fused QKV projection: per wave -> one (b, h, 16-row tile), full E=64.
// q,k stored [B,H,S,E]; v stored transposed [B,H,E,S] for the PV gemm.
// ---------------------------------------------------------------------------
__global__ __launch_bounds__(128) void qkv_kernel(
    const __bf16* __restrict__ xb, const __bf16* __restrict__ wqT,
    const __bf16* __restrict__ wkT, const __bf16* __restrict__ wvT,
    __bf16* __restrict__ q, __bf16* __restrict__ k, __bf16* __restrict__ vT) {
  const int w    = blockIdx.x * 4 + (threadIdx.x >> 5);
  const int lane = threadIdx.x & 31;
  const int g    = lane >> 4;   // lane-half
  const int lm   = lane & 15;   // row (A) / col (B,C)
  const int b    = w >> 11;           // 16*128 tiles per batch
  const int h    = (w >> 7) & 15;
  const int s0   = (w & 127) * 16;

  const __bf16* arow = xb + ((size_t)b * S_ + s0 + lm) * D_;
  const __bf16* bq   = wqT + (size_t)h * E_ * D_;
  const __bf16* bk   = wkT + (size_t)h * E_ * D_;
  const __bf16* bv   = wvT + (size_t)h * E_ * D_;

  f32x8 aq[4], ak[4], av[4];
#pragma unroll
  for (int j = 0; j < 4; ++j) { aq[j] = zero8(); ak[j] = zero8(); av[j] = zero8(); }

  for (int kb = 0; kb < D_; kb += 32) {
    bf16x16 a = a_frag(arow, kb, g);
#pragma unroll
    for (int j = 0; j < 4; ++j) {
      const size_t bro = (size_t)(j * 16 + lm) * D_ + kb + 16 * g;
      bf16x16 bb = *(const bf16x16*)(bq + bro);
      aq[j] = WMMA_BF16(a, bb, aq[j]);
      bb = *(const bf16x16*)(bk + bro);
      ak[j] = WMMA_BF16(a, bb, ak[j]);
      bb = *(const bf16x16*)(bv + bro);
      av[j] = WMMA_BF16(a, bb, av[j]);
    }
  }

  const size_t qbase = ((size_t)(b * H_ + h) * S_ + s0) * E_;
#pragma unroll
  for (int j = 0; j < 4; ++j) {
#pragma unroll
    for (int r = 0; r < 8; ++r) {
      const int row = r + 8 * g;
      q[qbase + (size_t)row * E_ + j * 16 + lm] = tobf(aq[j][r]);
      k[qbase + (size_t)row * E_ + j * 16 + lm] = tobf(ak[j][r]);
    }
  }
  // V transposed: rows r..r+7 of one column are contiguous in S -> pack 16B store
  const size_t vbase = (size_t)(b * H_ + h) * E_ * S_;
#pragma unroll
  for (int j = 0; j < 4; ++j) {
    bf16x8 pk;
#pragma unroll
    for (int r = 0; r < 8; ++r) pk[r] = tobf(av[j][r]);
    *(bf16x8*)(vT + vbase + (size_t)(j * 16 + lm) * S_ + s0 + 8 * g) = pk;
  }
}

// ---------------------------------------------------------------------------
// Flash attention (causal): per wave -> one (b, h, 16-row block).
// Streaming softmax in exp2 domain; P converted C-layout -> A-layout via LDS.
// Output written in concat-head layout [B,S,H*E] (bf16).
// ---------------------------------------------------------------------------
__global__ __launch_bounds__(128) void flash_kernel(
    const __bf16* __restrict__ q, const __bf16* __restrict__ k,
    const __bf16* __restrict__ vT, __bf16* __restrict__ ao) {
  __shared__ __align__(16) __bf16 plds[4][16][40];  // per-wave 16x32 P tile (padded)

  const int wid  = threadIdx.x >> 5;
  const int w    = blockIdx.x * 4 + wid;
  const int lane = threadIdx.x & 31;
  const int g    = lane >> 4;
  const int lm   = lane & 15;
  const int b    = w >> 11;
  const int h    = (w >> 7) & 15;
  const int rb   = w & 127;

  const __bf16* Q = q  + (size_t)(b * H_ + h) * S_ * E_;
  const __bf16* K = k  + (size_t)(b * H_ + h) * S_ * E_;
  const __bf16* V = vT + (size_t)(b * H_ + h) * E_ * S_;

  bf16x16 qa[2];
#pragma unroll
  for (int c = 0; c < 2; ++c) qa[c] = a_frag(Q + (size_t)(rb * 16 + lm) * E_, c * 32, g);

  f32x8 acc[4];
#pragma unroll
  for (int j = 0; j < 4; ++j) acc[j] = zero8();
  float mx[8], sm[8];
#pragma unroll
  for (int r = 0; r < 8; ++r) { mx[r] = -1e30f; sm[r] = 0.0f; }

  const float sc = 0.18033688011112042f;  // (1/sqrt(64)) * log2(e)
  const int nkb = (rb * 16 + 47) >> 5;    // key blocks of 32 covering cols <= row

  for (int tb = 0; tb < nkb; ++tb) {
    const int t0 = tb * 32;
    // ---- S = Q·Kᵀ : two 16-wide score tiles, K-dim = E = 64 (2 chunks) ----
    f32x8 s0v = zero8(), s1v = zero8();
#pragma unroll
    for (int c = 0; c < 2; ++c) {
      bf16x16 b0 = *(const bf16x16*)(K + (size_t)(t0 + lm) * E_ + c * 32 + 16 * g);
      bf16x16 b1 = *(const bf16x16*)(K + (size_t)(t0 + 16 + lm) * E_ + c * 32 + 16 * g);
      s0v = WMMA_BF16(qa[c], b0, s0v);
      s1v = WMMA_BF16(qa[c], b1, s1v);
    }
    // ---- masked streaming softmax (C layout: row = r + 8g, col = lm) ----
#pragma unroll
    for (int r = 0; r < 8; ++r) {
      const int row = rb * 16 + r + 8 * g;
      float a0 = (t0 + lm      <= row) ? s0v[r] * sc : -1e30f;
      float a1 = (t0 + 16 + lm <= row) ? s1v[r] * sc : -1e30f;
      float vmax = fmaxf(a0, a1);
      vmax = fmaxf(vmax, __shfl_xor(vmax, 8, 32));
      vmax = fmaxf(vmax, __shfl_xor(vmax, 4, 32));
      vmax = fmaxf(vmax, __shfl_xor(vmax, 2, 32));
      vmax = fmaxf(vmax, __shfl_xor(vmax, 1, 32));
      const float nm    = fmaxf(mx[r], vmax);
      const float alpha = exp2f(mx[r] - nm);
      const float p0 = exp2f(a0 - nm);
      const float p1 = exp2f(a1 - nm);
      float rs = p0 + p1;
      rs += __shfl_xor(rs, 8, 32);
      rs += __shfl_xor(rs, 4, 32);
      rs += __shfl_xor(rs, 2, 32);
      rs += __shfl_xor(rs, 1, 32);
      sm[r] = sm[r] * alpha + rs;
      mx[r] = nm;
      plds[wid][r + 8 * g][lm]      = tobf(p0);
      plds[wid][r + 8 * g][16 + lm] = tobf(p1);
#pragma unroll
      for (int j = 0; j < 4; ++j) acc[j][r] *= alpha;  // online rescale of O
    }
    // ---- P (A layout from LDS) · V (Bᵀ rows contiguous in S) ----
    bf16x8 plo = *(const bf16x8*)&plds[wid][lm][16 * g];
    bf16x8 phi = *(const bf16x8*)&plds[wid][lm][16 * g + 8];
    bf16x16 pa;
#pragma unroll
    for (int i = 0; i < 8; ++i) { pa[i] = plo[i]; pa[i + 8] = phi[i]; }
#pragma unroll
    for (int j = 0; j < 4; ++j) {
      bf16x16 bv = *(const bf16x16*)(V + (size_t)(j * 16 + lm) * S_ + t0 + 16 * g);
      acc[j] = WMMA_BF16(pa, bv, acc[j]);
    }
  }

  // ---- normalize and store concat-head output [B,S,H*E] ----
  const size_t obase = (size_t)b * S_ * D_ + (size_t)h * E_;
#pragma unroll
  for (int r = 0; r < 8; ++r) {
    const float inv = 1.0f / sm[r];
    const int row = rb * 16 + r + 8 * g;
#pragma unroll
    for (int j = 0; j < 4; ++j)
      ao[obase + (size_t)row * D_ + j * 16 + lm] = tobf(acc[j][r] * inv);
  }
}

// ---------------------------------------------------------------------------
// Output projection: out[m][n] = sum_d ao[m][d] * Wproj[n][d], fp32 out.
// Per wave: 16 rows x 64 cols.
// ---------------------------------------------------------------------------
__global__ __launch_bounds__(128) void proj_kernel(
    const __bf16* __restrict__ ao, const __bf16* __restrict__ wp,
    float* __restrict__ out) {
  const int w    = blockIdx.x * 4 + (threadIdx.x >> 5);
  const int lane = threadIdx.x & 31;
  const int g    = lane >> 4;
  const int lm   = lane & 15;
  const int mt   = w >> 4;   // 256 row tiles (B*S/16)
  const int nc   = w & 15;   // 16 col groups of 64

  const __bf16* arow = ao + (size_t)(mt * 16 + lm) * D_;
  f32x8 acc[4];
#pragma unroll
  for (int j = 0; j < 4; ++j) acc[j] = zero8();

  for (int kb = 0; kb < D_; kb += 32) {
    bf16x16 a = a_frag(arow, kb, g);
#pragma unroll
    for (int j = 0; j < 4; ++j) {
      bf16x16 bb = *(const bf16x16*)(wp + (size_t)(nc * 64 + j * 16 + lm) * D_ + kb + 16 * g);
      acc[j] = WMMA_BF16(a, bb, acc[j]);
    }
  }
#pragma unroll
  for (int r = 0; r < 8; ++r) {
#pragma unroll
    for (int j = 0; j < 4; ++j)
      out[(size_t)(mt * 16 + r + 8 * g) * D_ + nc * 64 + j * 16 + lm] = acc[j][r];
  }
}

// ---------------------------------------------------------------------------
// Host launcher
// ---------------------------------------------------------------------------
extern "C" void kernel_launch(void* const* d_in, const int* in_sizes, int n_in,
                              void* d_out, int out_size, void* d_ws, size_t ws_size,
                              hipStream_t stream) {
  (void)in_sizes; (void)n_in; (void)out_size; (void)ws_size;
  const float* x  = (const float*)d_in[0];
  const float* Wq = (const float*)d_in[1];
  const float* Wk = (const float*)d_in[2];
  const float* Wv = (const float*)d_in[3];
  const float* Wp = (const float*)d_in[4];
  float* out = (float*)d_out;

  const size_t BSD = (size_t)B_ * S_ * D_;        // 4,194,304
  const size_t HDE = (size_t)H_ * D_ * E_;        // 1,048,576
  const size_t DD  = (size_t)D_ * D_;             // 1,048,576
  const size_t BHSE = (size_t)B_ * H_ * S_ * E_;  // 4,194,304

  char* ws = (char*)d_ws;
  size_t off = 0;
  auto carve = [&](size_t bytes) -> void* {
    void* p = ws + off;
    off = (off + bytes + 255) & ~(size_t)255;
    return p;
  };
  __bf16* xb  = (__bf16*)carve(BSD * 2);
  __bf16* wqT = (__bf16*)carve(HDE * 2);
  __bf16* wkT = (__bf16*)carve(HDE * 2);
  __bf16* wvT = (__bf16*)carve(HDE * 2);
  __bf16* wpb = (__bf16*)carve(DD * 2);
  __bf16* qb  = (__bf16*)carve(BHSE * 2);
  __bf16* kb  = (__bf16*)carve(BHSE * 2);
  __bf16* vtb = (__bf16*)carve(BHSE * 2);
  __bf16* aob = (__bf16*)carve(BSD * 2);

  cvt_x_kernel<<<(int)((BSD + 255) / 256), 256, 0, stream>>>(x, xb, (int)BSD);
  cvt_w_kernel<<<(int)((HDE + 255) / 256), 256, 0, stream>>>(Wq, Wk, Wv, wqT, wkT, wvT);
  cvt_wp_kernel<<<(int)((DD + 255) / 256), 256, 0, stream>>>(Wp, wpb, (int)DD);

  // 4096 waves each: B*H*(S/16) tiles, 4 waves per 128-thread block
  qkv_kernel<<<1024, 128, 0, stream>>>(xb, wqT, wkT, wvT, qb, kb, vtb);
  flash_kernel<<<1024, 128, 0, stream>>>(qb, kb, vtb, aob);
  proj_kernel<<<1024, 128, 0, stream>>>(aob, wpb, out);
}
